// SectorAttention_2628519985691
// MI455X (gfx1250) — matrix operand order repaired and checked by reading, not compiled, not run on hardware
//
#include <hip/hip_runtime.h>
#include <math.h>

// ---------------------------------------------------------------------------
// SectorAttention forward for MI455X (gfx1250, wave32, WMMA).
// GEMMs run on v_wmma_f32_16x16x32_bf16 (fp32 data converted to bf16 in the
// LDS loaders, f32 accumulation).  Attention / LayerNorm / dynamic-bias net
// are cheap VALU kernels.
//
// Problem constants (from reference): B=4, Hq=4, Hres=4, Wres=4096, C=256,
// NH=8, HD=32, H_SP=4, W_SP=1.  Tokens M = B*Hq*Wres = 65536.
// Workspace requirement: 276,824,064 bytes (~264 MiB).
// ---------------------------------------------------------------------------

typedef __bf16 bf16x8  __attribute__((ext_vector_type(8)));
typedef __bf16 bf16x16 __attribute__((ext_vector_type(16)));
typedef float  v8f     __attribute__((ext_vector_type(8)));

union FragBF16 { bf16x16 v; bf16x8 h[2]; };

__device__ __forceinline__ unsigned short f2b(float f) {
    union { float f; unsigned u; } x; x.f = f;
    unsigned r = x.u + 0x7FFFu + ((x.u >> 16) & 1u);   // round-to-nearest-even
    return (unsigned short)(r >> 16);
}

__device__ __forceinline__ float gelu_exact(float x) {
    return 0.5f * x * (1.0f + erff(x * 0.70710678118654752f));
}

__device__ __forceinline__ float wave_sum32(float v) {
    #pragma unroll
    for (int off = 16; off > 0; off >>= 1) v += __shfl_xor(v, off, 32);
    return v;
}

// ---------------------------------------------------------------------------
// WMMA GEMM:  Out[M,N] = A[M,K] @ W[N,K]^T + bias[N]  (+ res / gelu)
// Block tile 256(M) x 64(N), KC=64 chunks via LDS; 8 waves, each wave owns a
// 32(M) x 64(N) strip -> 8 accumulators.  Per 32-wide k-step: 2 A fragments +
// 4 B fragments are preloaded into registers (12 ds_load_b128 under one wait)
// and then 8 v_wmma_f32_16x16x32_bf16 issue back-to-back with B reuse.
// EPI: 0 = bias, 1 = bias + residual add, 2 = bias + exact GELU.
// ---------------------------------------------------------------------------
#define BM 256
#define BN 64
#define KC 64
#define LDA 72   // bf16 elems per LDS row (64 + 8 pad, keeps 16B alignment)
#define LDB 72

template <bool A_BF16, bool OUT_BF16, int EPI>
__global__ __launch_bounds__(256) void gemm_kernel(
    const void* __restrict__ Aptr, const float* __restrict__ W,
    const float* __restrict__ bias, const float* __restrict__ res,
    void* __restrict__ Out, int M, int N, int K)
{
    __shared__ unsigned short sA[BM * LDA];   // 36 KB
    __shared__ unsigned short sB[BN * LDB];   //  9 KB

    const int tid  = threadIdx.x;
    const int wave = tid >> 5;
    const int lane = tid & 31;
    const int m0   = blockIdx.x * BM;
    const int n0   = blockIdx.y * BN;

    v8f acc[2][4];
    #pragma unroll
    for (int i = 0; i < 2; ++i)
        #pragma unroll
        for (int j = 0; j < 4; ++j) acc[i][j] = (v8f)(0.0f);

    const int mrow0 = wave * 32 + (lane & 15);  // first A-fragment row in tile
    const int hl    = lane >> 4;                // half-wave select

    for (int kc = 0; kc < K; kc += KC) {
        // ---- stage A chunk (BM x KC) into LDS as bf16 ----
        if (A_BF16) {
            const unsigned short* A = (const unsigned short*)Aptr;
            #pragma unroll
            for (int i = 0; i < 8; ++i) {           // 2048 x 16B units
                int u   = tid + i * 256;
                int row = u >> 3;
                int c8  = (u & 7) * 8;
                uint4 d = *(const uint4*)(A + (size_t)(m0 + row) * K + kc + c8);
                *(uint4*)&sA[row * LDA + c8] = d;
            }
        } else {
            const float* A = (const float*)Aptr;
            #pragma unroll
            for (int i = 0; i < 16; ++i) {          // 4096 float4 units
                int u   = tid + i * 256;
                int row = u >> 4;
                int c4  = (u & 15) * 4;
                float4 d = *(const float4*)(A + (size_t)(m0 + row) * K + kc + c4);
                ushort4 h;
                h.x = f2b(d.x); h.y = f2b(d.y); h.z = f2b(d.z); h.w = f2b(d.w);
                *(ushort4*)&sA[row * LDA + c4] = h;
            }
        }
        // ---- stage W chunk (BN x KC) into LDS as bf16 ----
        #pragma unroll
        for (int i = 0; i < 4; ++i) {               // 1024 float4 units
            int u   = tid + i * 256;
            int row = u >> 4;
            int c4  = (u & 15) * 4;
            float4 d = *(const float4*)(W + (size_t)(n0 + row) * K + kc + c4);
            ushort4 h;
            h.x = f2b(d.x); h.y = f2b(d.y); h.z = f2b(d.z); h.w = f2b(d.w);
            *(ushort4*)&sB[row * LDB + c4] = h;
        }
        __syncthreads();

        // ---- compute: 2 k-steps of 32; preload all frags, then 8 WMMAs ----
        #pragma unroll
        for (int ks = 0; ks < KC; ks += 32) {
            // CDNA5 16-bit A 16x32 layout: half-wave 0 -> K {0..7,16..23},
            // half-wave 1 -> K {8..15,24..31}, row M = lane&15.
            FragBF16 a0, a1;
            a0.h[0] = *(const bf16x8*)&sA[mrow0 * LDA + ks + hl * 8];
            a0.h[1] = *(const bf16x8*)&sA[mrow0 * LDA + ks + 16 + hl * 8];
            a1.h[0] = *(const bf16x8*)&sA[(mrow0 + 16) * LDA + ks + hl * 8];
            a1.h[1] = *(const bf16x8*)&sA[(mrow0 + 16) * LDA + ks + 16 + hl * 8];
            // B 32x16 layout: same K order as A, one column per lane,
            // half-wave 0 -> K {0..7,16..23}, half-wave 1 -> K {8..15,24..31}.
            FragBF16 b[4];
            #pragma unroll
            for (int j = 0; j < 4; ++j) {
                int ncol = j * 16 + (lane & 15);
                b[j].h[0] = *(const bf16x8*)&sB[ncol * LDB + ks + hl * 8];
                b[j].h[1] = *(const bf16x8*)&sB[ncol * LDB + ks + 16 + hl * 8];
            }
            #pragma unroll
            for (int j = 0; j < 4; ++j)
                acc[0][j] = __builtin_amdgcn_wmma_f32_16x16x32_bf16(
                    false, a0.v, false, b[j].v, (short)0, acc[0][j], false, false);
            #pragma unroll
            for (int j = 0; j < 4; ++j)
                acc[1][j] = __builtin_amdgcn_wmma_f32_16x16x32_bf16(
                    false, a1.v, false, b[j].v, (short)0, acc[1][j], false, false);
        }
        __syncthreads();
    }

    // ---- epilogue: C/D layout -> VGPR r holds M = (lane>>4)*8 + r, N = lane&15
    #pragma unroll
    for (int i = 0; i < 2; ++i) {
        const int rbase = m0 + wave * 32 + i * 16 + hl * 8;
        #pragma unroll
        for (int j = 0; j < 4; ++j) {
            int col  = n0 + j * 16 + (lane & 15);
            float bs = bias[col];
            #pragma unroll
            for (int r = 0; r < 8; ++r) {
                size_t idx = (size_t)(rbase + r) * N + col;
                float v = acc[i][j][r] + bs;
                if (EPI == 1) v += res[idx];
                if (EPI == 2) v = gelu_exact(v);
                if (OUT_BF16) ((unsigned short*)Out)[idx] = f2b(v);
                else          ((float*)Out)[idx]          = v;
            }
        }
    }
}

// ---------------------------------------------------------------------------
// Dynamic relative-position bias net: per window (nW=16384) a 2->16->8 conv
// (k=1) with eval-mode BatchNorm + ReLU over 16 positions. 1 thread / window.
// rel_bias layout: [w][head][hq][hs]  (w = b*4096 + ww)
// ---------------------------------------------------------------------------
__global__ __launch_bounds__(256) void bias_kernel(
    const float* __restrict__ s_pos, const float* __restrict__ x_pos,
    const float* __restrict__ pw1, const float* __restrict__ pb1,
    const float* __restrict__ bn_g, const float* __restrict__ bn_b,
    const float* __restrict__ bn_m, const float* __restrict__ bn_v,
    const float* __restrict__ pw2, const float* __restrict__ pb2,
    float* __restrict__ rel_bias)
{
    int w  = blockIdx.x * 256 + threadIdx.x;   // 0..16383
    int b  = w >> 12;
    int ww = w & 4095;

    // sp[c][q]: faithful reinterpretation of transposed s_pos block
    float sp[2][4], xp[2][4];
    #pragma unroll
    for (int c = 0; c < 2; ++c)
        #pragma unroll
        for (int q = 0; q < 4; ++q) {
            int idx = c * 4 + q;
            sp[c][q] = s_pos[(((size_t)b * 2 + (idx & 1)) * 4096 + ww) * 4 + (idx >> 1)];
        }
    #pragma unroll
    for (int c = 0; c < 2; ++c)
        #pragma unroll
        for (int j = 0; j < 4; ++j)
            xp[c][j] = x_pos[((size_t)b * 16384 + (size_t)j * 4096 + ww) * 2 + c];

    for (int l = 0; l < 16; ++l) {
        int q = l >> 2, j = l & 3;
        float r0 = sp[0][q] - xp[0][j];
        float r1 = sp[1][q] - xp[1][j];
        float o8[8];
        #pragma unroll
        for (int n = 0; n < 8; ++n) o8[n] = pb2[n];
        #pragma unroll
        for (int o = 0; o < 16; ++o) {
            float h = pw1[o * 2] * r0 + pw1[o * 2 + 1] * r1 + pb1[o];
            h = (h - bn_m[o]) * rsqrtf(bn_v[o] + 1e-5f) * bn_g[o] + bn_b[o];
            h = fmaxf(h, 0.0f);
            #pragma unroll
            for (int n = 0; n < 8; ++n) o8[n] += pw2[n * 16 + o] * h;
        }
        #pragma unroll
        for (int n = 0; n < 8; ++n)
            rel_bias[((size_t)w * 8 + n) * 16 + l] = o8[n];
    }
}

// ---------------------------------------------------------------------------
// Windowed attention: block = one window (8 waves = 8 heads), lane = head dim.
// q comes from the torch-.view reinterpretation of qproj, k/v are natural.
// 4x4 logits via cross-lane butterfly reductions, softmax + dynamic bias,
// per-lane AV.  Output written in windows2img layout -> o-proj is plain GEMM.
// Strided q reads hit L2 (qproj = 67 MB << 192 MB global L2).
// ---------------------------------------------------------------------------
__global__ __launch_bounds__(256) void attn_kernel(
    const float* __restrict__ qproj, const float* __restrict__ kproj,
    const float* __restrict__ vproj, const float* __restrict__ rel_bias,
    float* __restrict__ o_img)
{
    const int w  = blockIdx.x;            // window 0..16383
    const int n  = threadIdx.x >> 5;      // head 0..7
    const int d  = threadIdx.x & 31;      // head dim 0..31
    const int b  = w >> 12;
    const int ww = w & 4095;

    float q[4], k[4], v[4];
    // q[w][n][hq][d] = qproj flat at ((b*256 + n*32 + d)*4 + hq)*4096 + ww
    size_t qb = ((size_t)(b * 256 + n * 32 + d) * 4) * 4096 + ww;
    #pragma unroll
    for (int hq = 0; hq < 4; ++hq) q[hq] = qproj[qb + (size_t)hq * 4096];
    // k/v[w][n][hs][d] = proj[b, hs*4096+ww, n*32+d]
    size_t kb = ((size_t)b * 16384 + ww) * 256 + n * 32 + d;
    #pragma unroll
    for (int hs = 0; hs < 4; ++hs) {
        k[hs] = kproj[kb + (size_t)hs * 4096 * 256];
        v[hs] = vproj[kb + (size_t)hs * 4096 * 256];
    }

    const float scale = 0.17677669529663687f;   // 32^-0.5
    const float* rb = rel_bias + ((size_t)w * 8 + n) * 16;

    float att[16];
    #pragma unroll
    for (int hq = 0; hq < 4; ++hq)
        #pragma unroll
        for (int hs = 0; hs < 4; ++hs)
            att[hq * 4 + hs] = q[hq] * k[hs];
    #pragma unroll
    for (int i = 0; i < 16; ++i)
        att[i] = wave_sum32(att[i]) * scale + rb[i];

    #pragma unroll
    for (int hq = 0; hq < 4; ++hq) {
        float m = fmaxf(fmaxf(att[hq * 4], att[hq * 4 + 1]),
                        fmaxf(att[hq * 4 + 2], att[hq * 4 + 3]));
        float e0 = __expf(att[hq * 4 + 0] - m);
        float e1 = __expf(att[hq * 4 + 1] - m);
        float e2 = __expf(att[hq * 4 + 2] - m);
        float e3 = __expf(att[hq * 4 + 3] - m);
        float rs = 1.0f / (e0 + e1 + e2 + e3);
        float o = (e0 * v[0] + e1 * v[1] + e2 * v[2] + e3 * v[3]) * rs;
        o_img[((size_t)b * 16384 + (size_t)hq * 4096 + ww) * 256 + n * 32 + d] = o;
    }
}

// ---------------------------------------------------------------------------
// LayerNorm over C=256: one wave per row (8 values per lane).
// ---------------------------------------------------------------------------
__global__ __launch_bounds__(256) void ln_kernel(
    const float* __restrict__ s1, const float* __restrict__ g,
    const float* __restrict__ beta, float* __restrict__ outn)
{
    const int row  = blockIdx.x * 8 + (threadIdx.x >> 5);
    const int lane = threadIdx.x & 31;
    const float* xr = s1 + (size_t)row * 256 + lane * 8;
    float4 a = *(const float4*)(xr);
    float4 c = *(const float4*)(xr + 4);
    float x[8] = {a.x, a.y, a.z, a.w, c.x, c.y, c.z, c.w};

    float s = 0.f;
    #pragma unroll
    for (int j = 0; j < 8; ++j) s += x[j];
    float mu = wave_sum32(s) * (1.0f / 256.0f);
    float vs = 0.f;
    #pragma unroll
    for (int j = 0; j < 8; ++j) { float t = x[j] - mu; vs += t * t; }
    float rstd = rsqrtf(wave_sum32(vs) * (1.0f / 256.0f) + 1e-5f);

    float y[8];
    #pragma unroll
    for (int j = 0; j < 8; ++j)
        y[j] = (x[j] - mu) * rstd * g[lane * 8 + j] + beta[lane * 8 + j];
    float* orow = outn + (size_t)row * 256 + lane * 8;
    *(float4*)(orow)     = make_float4(y[0], y[1], y[2], y[3]);
    *(float4*)(orow + 4) = make_float4(y[4], y[5], y[6], y[7]);
}

// ---------------------------------------------------------------------------
extern "C" void kernel_launch(void* const* d_in, const int* in_sizes, int n_in,
                              void* d_out, int out_size, void* d_ws, size_t ws_size,
                              hipStream_t stream) {
    const float* s    = (const float*)d_in[0];
    const float* x    = (const float*)d_in[1];
    const float* spos = (const float*)d_in[2];
    const float* xpos = (const float*)d_in[3];
    const float* Wq = (const float*)d_in[4];  const float* bq = (const float*)d_in[5];
    const float* Wk = (const float*)d_in[6];  const float* bk = (const float*)d_in[7];
    const float* Wv = (const float*)d_in[8];  const float* bv = (const float*)d_in[9];
    const float* Wo = (const float*)d_in[10]; const float* bo = (const float*)d_in[11];
    const float* pw1 = (const float*)d_in[12]; const float* pb1 = (const float*)d_in[13];
    const float* bng = (const float*)d_in[14]; const float* bnb = (const float*)d_in[15];
    const float* bnm = (const float*)d_in[16]; const float* bnv = (const float*)d_in[17];
    const float* pw2 = (const float*)d_in[18]; const float* pb2 = (const float*)d_in[19];
    const float* lng = (const float*)d_in[20]; const float* lnb = (const float*)d_in[21];
    const float* mw1 = (const float*)d_in[22]; const float* mb1 = (const float*)d_in[23];
    const float* mw2 = (const float*)d_in[24]; const float* mb2 = (const float*)d_in[25];

    // Workspace carving with liveness reuse (total 276,824,064 B ~ 264 MiB):
    //   [0,   64M)  qproj      -> reused as s1
    //   [64M, 128M) kproj      -> reused as n (LayerNorm out)
    //   [128M,192M) vproj      -> reused (with relb+o_img span) as bf16 h2
    //   [192M,200M) rel_bias
    //   [200M,264M) o_img
    char* ws = (char*)d_ws;
    float*          qproj = (float*)(ws + 0);
    float*          kproj = (float*)(ws + 67108864);
    float*          vproj = (float*)(ws + 134217728);
    unsigned short* h2    = (unsigned short*)(ws + 134217728);  // after vproj dies
    float*          relb  = (float*)(ws + 201326592);
    float*          o_img = (float*)(ws + 209715200);
    float*          s1    = qproj;   // after qproj dies
    float*          nbuf  = kproj;   // after kproj dies

    const int M = 65536;             // B * Hq * Wres tokens
    dim3 blk(256);
    dim3 gN256(M / BM, 256 / BN);    // (256, 4)
    dim3 gN1024(M / BM, 1024 / BN);  // (256, 16)

    bias_kernel<<<16384 / 256, blk, 0, stream>>>(spos, xpos, pw1, pb1, bng, bnb,
                                                 bnm, bnv, pw2, pb2, relb);
    gemm_kernel<false, false, 0><<<gN256, blk, 0, stream>>>(s, Wq, bq, nullptr, qproj, M, 256, 256);
    gemm_kernel<false, false, 0><<<gN256, blk, 0, stream>>>(x, Wk, bk, nullptr, kproj, M, 256, 256);
    gemm_kernel<false, false, 0><<<gN256, blk, 0, stream>>>(x, Wv, bv, nullptr, vproj, M, 256, 256);
    attn_kernel<<<16384, blk, 0, stream>>>(qproj, kproj, vproj, relb, o_img);
    gemm_kernel<false, false, 1><<<gN256, blk, 0, stream>>>(o_img, Wo, bo, s, s1, M, 256, 256);
    ln_kernel<<<M / 8, blk, 0, stream>>>(s1, lng, lnb, nbuf);
    gemm_kernel<false, true, 2><<<gN1024, blk, 0, stream>>>(nbuf, mw1, mb1, nullptr, h2, M, 1024, 256);
    gemm_kernel<true, false, 1><<<gN256, blk, 0, stream>>>(h2, mw2, mb2, s1, (float*)d_out, M, 256, 1024);
}
